// MultiScaleAttention_61649960567111
// MI455X (gfx1250) — matrix operand order repaired and checked by reading, not compiled, not run on hardware
//
#include <hip/hip_runtime.h>

// ---------------------------------------------------------------------------
// MultiScaleAttention for MI455X (gfx1250, wave32, WMMA 16x16x32 f16)
// ---------------------------------------------------------------------------
//   1) f32 -> f16 conversion of x, wq, wk, wv, wo
//   2) fused QKV GEMM (NT), double-buffered TDM (tensor_load_to_lds) staging
//   3) conv1d over L (scales 1/3/5 fused, LDS row staging) -> Vs1/2/3 f16
//   4) flash attention: scores once per tile, applied to 3 V tensors;
//      V tiles staged via global_load_async_to_lds_b128 (ASYNCcnt)
//   5) final NT GEMM + bias -> f32 out
// ---------------------------------------------------------------------------

typedef _Float16 half_t;
typedef _Float16 v16h __attribute__((ext_vector_type(16)));
typedef float    v8f  __attribute__((ext_vector_type(8)));
typedef float    vf4  __attribute__((ext_vector_type(4)));
typedef unsigned v4u  __attribute__((ext_vector_type(4)));
typedef int      v8i  __attribute__((ext_vector_type(8)));
typedef int      v4i  __attribute__((ext_vector_type(4)));
typedef unsigned u32;
typedef unsigned long long u64;

#define B_   2
#define L_   2048
#define D_   1024
#define H_   16
#define DH_  64
#define M_   (B_ * L_)     // 4096 rows
#define N3_  (3 * D_)      // 3072

union Frag {
    v16h   v;
    u32    u[8];
    half_t h[16];
    vf4    f4[2];
};

static __device__ inline v8f wmma_f16(v16h a, v16h b, v8f c) {
    return __builtin_amdgcn_wmma_f32_16x16x32_f16(
        false, a, false, b, (short)0, c, false, false);
}

// --- per-lane async global->LDS (ASYNCcnt) --------------------------------
static __device__ inline void async_row128(unsigned lds, const half_t* g) {
    asm volatile(
        "global_load_async_to_lds_b128 %0, %1, off\n\t"
        "global_load_async_to_lds_b128 %0, %1, off offset:16\n\t"
        "global_load_async_to_lds_b128 %0, %1, off offset:32\n\t"
        "global_load_async_to_lds_b128 %0, %1, off offset:48\n\t"
        "global_load_async_to_lds_b128 %0, %1, off offset:64\n\t"
        "global_load_async_to_lds_b128 %0, %1, off offset:80\n\t"
        "global_load_async_to_lds_b128 %0, %1, off offset:96\n\t"
        "global_load_async_to_lds_b128 %0, %1, off offset:112"
        :: "v"(lds), "v"((u64)g) : "memory");
}
static __device__ inline void wait_async0() {
    asm volatile("s_wait_asynccnt 0x0" ::: "memory");
}

// --- TDM: one 2-D tile descriptor, rows x cols f16 elements ----------------
// D# group0/group1 packed per CDNA5 ISA 8.3/8.4 (2-D tensor, groups 2/3 zero)
static __device__ inline void tdm_load_2d(unsigned lds_addr, const void* gaddr,
                                          unsigned tdim0 /*row len, elems*/,
                                          unsigned rows, unsigned cols,
                                          unsigned stride0 /*elems*/) {
    const u64 ga = (u64)gaddr;
    v4u g0;
    g0.x = 1u;                                   // count=1, user mode
    g0.y = lds_addr;                             // lds byte address
    g0.z = (u32)(ga & 0xFFFFFFFFu);              // global_addr[31:0]
    g0.w = (u32)((ga >> 32) & 0x1FFFFFFu)        // global_addr[56:32]
         | (2u << 30);                           // type = 2 (image)
    v8i g1;
    g1[0] = (int)(1u << 16);                     // data_size=1 -> 2 bytes
    g1[1] = (int)((tdim0 & 0xFFFFu) << 16);      // tensor_dim0 lo
    g1[2] = (int)((tdim0 >> 16) | ((rows & 0xFFFFu) << 16));  // dim0 hi|dim1 lo
    g1[3] = (int)((rows >> 16) | (cols << 16));  // dim1 hi | tile_dim0
    g1[4] = (int)(rows & 0xFFFFu);               // tile_dim1 (tile_dim2 = 0)
    g1[5] = (int)stride0;                        // tensor_dim0_stride[31:0]
    g1[6] = 0;
    g1[7] = 0;
    const v4i z4 = {0, 0, 0, 0};
    const v8i z8 = {0, 0, 0, 0, 0, 0, 0, 0};
    // amdgpu-toolchain (clang-23) 6-arg form
    __builtin_amdgcn_tensor_load_to_lds(g0, g1, z4, z4, z8, 0);
}

// ---------------------------------------------------------------------------
// 1) f32 -> f16 elementwise convert
// ---------------------------------------------------------------------------
__global__ void cvt_f32_f16(const float* __restrict__ in,
                            half_t* __restrict__ out, int n) {
    int i = blockIdx.x * blockDim.x + threadIdx.x;
    if (i < n) out[i] = (half_t)in[i];
}

// ---------------------------------------------------------------------------
// 2)+(5) NT GEMM: C[M,N] = A[M,K] * W[N,K]^T  (f16 in, f32 accum)
//    256 thr / 8 waves; block tile 128x64; wave tile 32x32;
//    double-buffered TDM staging (wave 0 issues descriptors, TENSORcnt).
// ---------------------------------------------------------------------------
__global__ __launch_bounds__(256)
void gemm_nt(const half_t* __restrict__ A,
             const half_t* __restrict__ Bq, const half_t* __restrict__ Bk,
             const half_t* __restrict__ Bv,
             half_t* __restrict__ Oq, half_t* __restrict__ Ok,
             half_t* __restrict__ Ov,
             const float* __restrict__ bias, float* __restrict__ Of32,
             int Kdim, int mode) {
    __shared__ __align__(16) half_t As[2][128 * 32];   // 16 KB
    __shared__ __align__(16) half_t Bs[2][64 * 32];    //  8 KB

    const half_t* Bmat = Bq;
    half_t* Oh = Oq;
    if (mode == 0) {
        Bmat = (blockIdx.z == 0) ? Bq : (blockIdx.z == 1) ? Bk : Bv;
        Oh   = (blockIdx.z == 0) ? Oq : (blockIdx.z == 1) ? Ok : Ov;
    }

    const int tid = threadIdx.x;
    const int rowbase = blockIdx.x * 128;
    const int colbase = blockIdx.y * 64;
    const int lane = tid & 31, wid = tid >> 5;
    const int wrow = wid >> 1, wcol = wid & 1;
    const int hi = lane >> 4, ln = lane & 15;

    v8f acc[2][2] = {};

    auto stage = [&](int buf, int kb) {
        // A tile 128x32, B tile 64x32 — one TDM descriptor each
        tdm_load_2d((unsigned)(size_t)&As[buf][0],
                    A + (size_t)rowbase * Kdim + kb, (unsigned)Kdim,
                    128u, 32u, (unsigned)Kdim);
        tdm_load_2d((unsigned)(size_t)&Bs[buf][0],
                    Bmat + (size_t)colbase * Kdim + kb, (unsigned)Kdim,
                    64u, 32u, (unsigned)Kdim);
    };

    if (wid == 0) {
        stage(0, 0);
        __builtin_amdgcn_s_wait_tensorcnt(0);
    }
    __syncthreads();

    int buf = 0;
    for (int kb = 0; kb < Kdim; kb += 32) {
        if (wid == 0 && kb + 32 < Kdim) stage(buf ^ 1, kb + 32);

        Frag af[2], bf[2];
#pragma unroll
        for (int i = 0; i < 2; i++) {
            const int mrow = wrow * 32 + i * 16 + ln;
#pragma unroll
            for (int v = 0; v < 8; v++) {
                // A frag, contiguous k order used here (B matches): lanes0-15 K=0..15 ; lanes16-31 K=16..31
                const int k0 = 2 * v + hi * 16;
                af[i].u[v] = *(const u32*)(&As[buf][mrow * 32 + k0]);
            }
        }
#pragma unroll
        for (int j = 0; j < 2; j++) {
            // B frag, same k order as A: lane = N, halfs = contiguous K (+16 hi half)
            const int nrow = wcol * 32 + j * 16 + ln;
            const half_t* bp = &Bs[buf][nrow * 32 + hi * 16];
            bf[j].f4[0] = *(const vf4*)(bp);
            bf[j].f4[1] = *(const vf4*)(bp + 8);
        }
#pragma unroll
        for (int i = 0; i < 2; i++)
#pragma unroll
            for (int j = 0; j < 2; j++)
                acc[i][j] = wmma_f16(af[i].v, bf[j].v, acc[i][j]);

        if (wid == 0) __builtin_amdgcn_s_wait_tensorcnt(0);
        __syncthreads();
        buf ^= 1;
    }

    // epilogue (C layout: VGPR r -> M = r + 8*hi, N = ln)
#pragma unroll
    for (int i = 0; i < 2; i++) {
#pragma unroll
        for (int j = 0; j < 2; j++) {
            const int n = colbase + wcol * 32 + j * 16 + ln;
#pragma unroll
            for (int r = 0; r < 8; r++) {
                const int m = rowbase + wrow * 32 + i * 16 + hi * 8 + r;
                const float val = acc[i][j][r];
                if (mode == 0) {
                    const int bb = m >> 11, ll = m & (L_ - 1);
                    const int hh = n >> 6, dd = n & 63;
                    Oh[(((size_t)bb * H_ + hh) * L_ + ll) * DH_ + dd] = (half_t)val;
                } else {
                    Of32[(size_t)m * D_ + n] = val + bias[n];
                }
            }
        }
    }
}

// ---------------------------------------------------------------------------
// 3) conv1d along L, 3 scales fused. LDS-staged input rows, vectorized
//    weight loads (weights are L2-resident: ~100 KB vs 192 MB L2).
// ---------------------------------------------------------------------------
__global__ __launch_bounds__(64)
void conv_kernel(const half_t* __restrict__ V,
                 const float* __restrict__ w1, const float* __restrict__ b1,
                 const float* __restrict__ w3, const float* __restrict__ b3,
                 const float* __restrict__ w5, const float* __restrict__ b5,
                 half_t* __restrict__ Vs1, half_t* __restrict__ Vs2,
                 half_t* __restrict__ Vs3) {
    __shared__ __align__(16) half_t sv[5][DH_];   // 640 B: rows l-2 .. l+2
    const int d = threadIdx.x;
    const int idx = blockIdx.x;
    const int l = idx & (L_ - 1);
    const int bh = idx >> 11;
    const half_t* vb = V + (size_t)bh * L_ * DH_;

    // cooperative stage (zero padding folded in): 40 threads x 16B
    if (d < 40) {
        const int t = d >> 3, off = (d & 7) * 8;
        const int lr = l + t - 2;
        vf4 val = {0.f, 0.f, 0.f, 0.f};
        if (lr >= 0 && lr < L_) val = *(const vf4*)(vb + (size_t)lr * DH_ + off);
        *(vf4*)&sv[t][off] = val;
    }
    __syncthreads();

    float a1 = 0.f, a3 = 0.f, a5 = 0.f;
    const float* w5r = w5 + d * (DH_ * 5);
    const float* w3r = w3 + d * (DH_ * 3);
    const float* w1r = w1 + d * DH_;

    for (int cc = 0; cc < DH_; cc += 8) {
        union { vf4 f; half_t h[8]; } rv[5];
#pragma unroll
        for (int t = 0; t < 5; t++) rv[t].f = *(const vf4*)&sv[t][cc];
#pragma unroll
        for (int k = 0; k < 8; k++) {
            const int c = cc + k;
            const float v0 = (float)rv[0].h[k];
            const float v1 = (float)rv[1].h[k];
            const float v2 = (float)rv[2].h[k];
            const float v3 = (float)rv[3].h[k];
            const float v4 = (float)rv[4].h[k];
            a5 += v0 * w5r[c * 5 + 0] + v1 * w5r[c * 5 + 1] +
                  v2 * w5r[c * 5 + 2] + v3 * w5r[c * 5 + 3] +
                  v4 * w5r[c * 5 + 4];
            a3 += v1 * w3r[c * 3 + 0] + v2 * w3r[c * 3 + 1] +
                  v3 * w3r[c * 3 + 2];
            a1 += v2 * w1r[c];
        }
    }
    const size_t o = (size_t)bh * L_ * DH_ + (size_t)l * DH_ + d;
    Vs1[o] = (half_t)(a1 + b1[d]);
    Vs2[o] = (half_t)(a3 + b3[d]);
    Vs3[o] = (half_t)(a5 + b5[d]);
}

// ---------------------------------------------------------------------------
// 4) flash attention, 3 V tensors, per-wave 16-row Q block.
// ---------------------------------------------------------------------------
__global__ __launch_bounds__(128)
void flash_kernel(const half_t* __restrict__ Q, const half_t* __restrict__ K,
                  const half_t* __restrict__ V1, const half_t* __restrict__ V2,
                  const half_t* __restrict__ V3, half_t* __restrict__ Comb) {
    __shared__ __align__(16) float  sS[4][16][32];       //  8 KB
    __shared__ __align__(16) half_t sP[4][16][32];       //  4 KB
    __shared__ __align__(16) half_t sV[4][3][32 * 64];   // 48 KB
    __shared__ __align__(16) float  sMax[4][16];
    __shared__ __align__(16) float  sSum[4][16];
    __shared__ __align__(16) float  sFac[4][16];

    const int tid = threadIdx.x;
    const int wid = tid >> 5, lane = tid & 31;
    const int hi = lane >> 4, ln = lane & 15;
    const int bh = blockIdx.y;
    const int ib = (blockIdx.x * 4 + wid) * 16;

    const half_t* qp = Q + (size_t)bh * L_ * DH_;
    const half_t* kp = K + (size_t)bh * L_ * DH_;
    const half_t* vp[3] = {V1 + (size_t)bh * L_ * DH_,
                           V2 + (size_t)bh * L_ * DH_,
                           V3 + (size_t)bh * L_ * DH_};
    unsigned ldsV[3];
#pragma unroll
    for (int s = 0; s < 3; s++)
        ldsV[s] = (unsigned)(size_t)&sV[wid][s][lane * DH_];

    // Q fragments (A layout), two 32-wide K-chunks over dh=64
    Frag qf[2];
#pragma unroll
    for (int c = 0; c < 2; c++)
#pragma unroll
        for (int v = 0; v < 8; v++) {
            const int k0 = 2 * v + hi * 16 + c * 32;
            qf[c].u[v] = *(const u32*)(qp + (size_t)(ib + ln) * DH_ + k0);
        }

    v8f oacc[3][4] = {};
    if (hi == 0) { sMax[wid][ln] = -1e30f; sSum[wid][ln] = 0.f; }

    for (int jb = 0; jb < L_; jb += 32) {
        // kick off async V staging (overlaps S + softmax)
#pragma unroll
        for (int s = 0; s < 3; s++)
            async_row128(ldsV[s], vp[s] + (size_t)(jb + lane) * DH_);

        // S = (Q K^T) / sqrt(dh)
        v8f sacc[2];
#pragma unroll
        for (int jt = 0; jt < 2; jt++) {
            v8f a = {};
#pragma unroll
            for (int c = 0; c < 2; c++) {
                Frag kf;
                const int n = jb + jt * 16 + ln;
                const half_t* kr = kp + (size_t)n * DH_ + c * 32 + hi * 16;
                kf.f4[0] = *(const vf4*)(kr);
                kf.f4[1] = *(const vf4*)(kr + 8);
                a = wmma_f16(qf[c].v, kf.v, a);
            }
            sacc[jt] = a;
        }
        __builtin_prefetch(kp + (size_t)(jb + 32 + lane) * DH_, 0, 1);

#pragma unroll
        for (int jt = 0; jt < 2; jt++)
#pragma unroll
            for (int r = 0; r < 8; r++)
                sS[wid][r + hi * 8][jt * 16 + ln] = sacc[jt][r] * 0.125f;

        // online softmax (half-waves duplicate row ln; benign same-value races)
        float rowv[32];
#pragma unroll
        for (int c = 0; c < 8; c++) {
            const vf4 t = *(const vf4*)&sS[wid][ln][c * 4];
            rowv[c * 4 + 0] = t.x; rowv[c * 4 + 1] = t.y;
            rowv[c * 4 + 2] = t.z; rowv[c * 4 + 3] = t.w;
        }
        float tmax = -1e30f;
#pragma unroll
        for (int c = 0; c < 32; c++) tmax = fmaxf(tmax, rowv[c]);
        const float mo = sMax[wid][ln];
        const float mn = fmaxf(mo, tmax);
        const float fac = __expf(mo - mn);
        float tsum = 0.f;
#pragma unroll
        for (int c = 0; c < 32; c++) {
            const float p = __expf(rowv[c] - mn);
            tsum += p;
            sP[wid][ln][c] = (half_t)p;
        }
        sSum[wid][ln] = sSum[wid][ln] * fac + tsum;
        sMax[wid][ln] = mn;
        sFac[wid][ln] = fac;

        float fr[8];
#pragma unroll
        for (int r = 0; r < 8; r++) fr[r] = sFac[wid][hi * 8 + r];
#pragma unroll
        for (int s = 0; s < 3; s++)
#pragma unroll
            for (int t = 0; t < 4; t++)
#pragma unroll
                for (int r = 0; r < 8; r++) oacc[s][t][r] *= fr[r];

        Frag pf;
#pragma unroll
        for (int v = 0; v < 8; v++) {
            const int k0 = 2 * v + hi * 16;
            pf.u[v] = *(const u32*)&sP[wid][ln][k0];
        }

        wait_async0();   // V tiles landed

#pragma unroll
        for (int s = 0; s < 3; s++) {
#pragma unroll
            for (int t = 0; t < 4; t++) {
                Frag vf;
#pragma unroll
                for (int h = 0; h < 16; h++) {
                    const int j = h + hi * 16;
                    vf.h[h] = sV[wid][s][j * DH_ + t * 16 + ln];
                }
                oacc[s][t] = wmma_f16(pf.v, vf.v, oacc[s][t]);
            }
        }
    }

    float inv[8];
#pragma unroll
    for (int r = 0; r < 8; r++) inv[r] = 1.f / sSum[wid][hi * 8 + r];
    const int bb = bh >> 4, hh = bh & 15;
#pragma unroll
    for (int s = 0; s < 3; s++)
#pragma unroll
        for (int t = 0; t < 4; t++)
#pragma unroll
            for (int r = 0; r < 8; r++) {
                const int g = ib + hi * 8 + r;
                const int col = s * D_ + hh * DH_ + t * 16 + ln;
                Comb[((size_t)bb * L_ + g) * N3_ + col] =
                    (half_t)(oacc[s][t][r] * inv[r]);
            }
}

// ---------------------------------------------------------------------------
// Host launch
// ---------------------------------------------------------------------------
extern "C" void kernel_launch(void* const* d_in, const int* in_sizes, int n_in,
                              void* d_out, int out_size, void* d_ws,
                              size_t ws_size, hipStream_t stream) {
    const float* x   = (const float*)d_in[0];
    const float* wq  = (const float*)d_in[1];
    const float* wk  = (const float*)d_in[2];
    const float* wv  = (const float*)d_in[3];
    const float* wo  = (const float*)d_in[4];
    const float* bo  = (const float*)d_in[5];
    const float* cw1 = (const float*)d_in[6];
    const float* cb1 = (const float*)d_in[7];
    const float* cw3 = (const float*)d_in[8];
    const float* cb3 = (const float*)d_in[9];
    const float* cw5 = (const float*)d_in[10];
    const float* cb5 = (const float*)d_in[11];
    float* out = (float*)d_out;

    char* ws = (char*)d_ws;
    half_t* Xh  = (half_t*)ws; ws += (size_t)M_ * D_ * 2;
    half_t* Wqh = (half_t*)ws; ws += (size_t)D_ * D_ * 2;
    half_t* Wkh = (half_t*)ws; ws += (size_t)D_ * D_ * 2;
    half_t* Wvh = (half_t*)ws; ws += (size_t)D_ * D_ * 2;
    half_t* Woh = (half_t*)ws; ws += (size_t)D_ * N3_ * 2;
    half_t* Qh  = (half_t*)ws; ws += (size_t)M_ * D_ * 2;
    half_t* Kh  = (half_t*)ws; ws += (size_t)M_ * D_ * 2;
    half_t* Vh  = (half_t*)ws; ws += (size_t)M_ * D_ * 2;
    half_t* Vs1 = (half_t*)ws; ws += (size_t)M_ * D_ * 2;
    half_t* Vs2 = (half_t*)ws; ws += (size_t)M_ * D_ * 2;
    half_t* Vs3 = (half_t*)ws; ws += (size_t)M_ * D_ * 2;
    half_t* Comb = (half_t*)ws; ws += (size_t)M_ * N3_ * 2;

    {
        int n = M_ * D_;
        cvt_f32_f16<<<(n + 255) / 256, 256, 0, stream>>>(x, Xh, n);
        n = D_ * D_;
        cvt_f32_f16<<<(n + 255) / 256, 256, 0, stream>>>(wq, Wqh, n);
        cvt_f32_f16<<<(n + 255) / 256, 256, 0, stream>>>(wk, Wkh, n);
        cvt_f32_f16<<<(n + 255) / 256, 256, 0, stream>>>(wv, Wvh, n);
        n = D_ * N3_;
        cvt_f32_f16<<<(n + 255) / 256, 256, 0, stream>>>(wo, Woh, n);
    }
    {
        dim3 grid(M_ / 128, D_ / 64, 3);
        gemm_nt<<<grid, 256, 0, stream>>>(Xh, Wqh, Wkh, Wvh, Qh, Kh, Vh,
                                          nullptr, nullptr, D_, 0);
    }
    conv_kernel<<<B_ * H_ * L_, 64, 0, stream>>>(Vh, cw1, cb1, cw3, cb3, cw5,
                                                 cb5, Vs1, Vs2, Vs3);
    {
        dim3 grid(L_ / 16 / 4, B_ * H_);
        flash_kernel<<<grid, 128, 0, stream>>>(Qh, Kh, Vs1, Vs2, Vs3, Comb);
    }
    {
        dim3 grid(M_ / 128, D_ / 64, 1);
        gemm_nt<<<grid, 256, 0, stream>>>(Comb, Woh, nullptr, nullptr, nullptr,
                                          nullptr, nullptr, bo, out, N3_, 1);
    }
}
